// MultiHeadAttention_73813307949474
// MI455X (gfx1250) — compile-verified, matrix operand order checked by reading, not run on hardware
//
#include <hip/hip_runtime.h>

#define DM 768
#define HEADS 12
#define DH 64
#define BB 2
#define TT 4096
#define MTOT (BB * TT) /* 8192 */

typedef __attribute__((ext_vector_type(16))) __bf16        v16bf;
typedef __attribute__((ext_vector_type(8)))  float         v8f;
typedef __attribute__((ext_vector_type(4)))  unsigned int  v4u;

#define WMMA_BF16(A, B, C) \
  __builtin_amdgcn_wmma_f32_16x16x32_bf16(false, (A), false, (B), (short)0, (C), false, false)

struct Frag { union { v16bf v; v4u q[2]; }; };

__device__ __forceinline__ unsigned short f2bf(float f) {
  unsigned int u = __float_as_uint(f);
  u += 0x7FFFu + ((u >> 16) & 1u);     // round-to-nearest-even
  return (unsigned short)(u >> 16);
}

__device__ __forceinline__ v8f vzero8() {
  v8f v;
#pragma unroll
  for (int i = 0; i < 8; ++i) v[i] = 0.f;
  return v;
}

// NT fragment loader from global: lane(0-15)=row l; K chunks at k0+8*half+{0..7},+16.
__device__ __forceinline__ Frag gfrag(const unsigned short* base, int ld,
                                      int row0, int k0, int lane) {
  Frag f;
  const unsigned short* p =
      base + (size_t)(row0 + (lane & 15)) * ld + (size_t)(k0 + ((lane >> 4) << 3));
  f.q[0] = *(const v4u*)p;
  f.q[1] = *(const v4u*)(p + 16);
  return f;
}

// Same fragment pattern but from an LDS tile with padded row stride (ushorts).
__device__ __forceinline__ Frag lfrag(const unsigned short* base, int rstride,
                                      int row0, int k0, int lane) {
  Frag f;
  const unsigned short* p =
      base + (row0 + (lane & 15)) * rstride + k0 + ((lane >> 4) << 3);
  f.q[0] = *(const v4u*)p;
  f.q[1] = *(const v4u*)(p + 16);
  return f;
}

// CDNA5 async copy: 16B global -> LDS, tracked by ASYNCcnt.
__device__ __forceinline__ void async_b128(unsigned lds_off, const void* g) {
  asm volatile("global_load_async_to_lds_b128 %0, %1, off"
               :: "v"(lds_off), "v"((unsigned long long)(size_t)g) : "memory");
}

// ---------------- fp32 -> bf16 bulk convert ----------------
__global__ void cvt_kernel(const float* __restrict__ src,
                           unsigned short* __restrict__ dst, int n) {
  int i = blockIdx.x * blockDim.x + threadIdx.x;
  if (i < n) dst[i] = f2bf(src[i]);
}

// ------------- weight transpose + convert (LDS tiled) -------------
__global__ void wtrans_kernel(const float* __restrict__ src,
                              unsigned short* __restrict__ dst) {
  __shared__ float tile[16][17];
  int n = blockIdx.x * 16 + threadIdx.x;
  int k = blockIdx.y * 16 + threadIdx.y;
  tile[threadIdx.y][threadIdx.x] = src[(size_t)k * DM + n];
  __syncthreads();
  int n2 = blockIdx.x * 16 + threadIdx.y;
  int k2 = blockIdx.y * 16 + threadIdx.x;
  dst[(size_t)n2 * DM + k2] = f2bf(tile[threadIdx.x][threadIdx.y]);
}

// ---------------- QKV projection: NT GEMM, register double-buffered ----------------
__global__ __launch_bounds__(128) void qkv_gemm_kernel(
    const unsigned short* __restrict__ xb,
    const unsigned short* __restrict__ wqT, const unsigned short* __restrict__ wkT,
    const unsigned short* __restrict__ wvT,
    const float* __restrict__ bq, const float* __restrict__ bk,
    const float* __restrict__ bv,
    unsigned short* __restrict__ qh, unsigned short* __restrict__ kh,
    unsigned short* __restrict__ vt) {
  const int lane = threadIdx.x & 31;
  const int wid  = threadIdx.x >> 5;
  const int m0   = blockIdx.x * 64 + wid * 16;
  const int n0   = blockIdx.y * 64;
  const int z    = blockIdx.z;
  const unsigned short* W = (z == 0) ? wqT : (z == 1) ? wkT : wvT;
  const float* bias       = (z == 0) ? bq  : (z == 1) ? bk  : bv;

  v8f acc[4];
#pragma unroll
  for (int nt = 0; nt < 4; ++nt) acc[nt] = vzero8();

  Frag a = gfrag(xb, DM, m0, 0, lane);
  Frag b[4];
#pragma unroll
  for (int nt = 0; nt < 4; ++nt) b[nt] = gfrag(W, DM, n0 + nt * 16, 0, lane);

#pragma unroll 4
  for (int k = 0; k < DM; k += 32) {
    int kn = (k + 32 < DM) ? (k + 32) : 0;     // wrap: dead prefetch on last iter
    Frag a2 = gfrag(xb, DM, m0, kn, lane);
    Frag b2[4];
#pragma unroll
    for (int nt = 0; nt < 4; ++nt) b2[nt] = gfrag(W, DM, n0 + nt * 16, kn, lane);
#pragma unroll
    for (int nt = 0; nt < 4; ++nt) acc[nt] = WMMA_BF16(a.v, b[nt].v, acc[nt]);
    a = a2;
#pragma unroll
    for (int nt = 0; nt < 4; ++nt) b[nt] = b2[nt];
  }

  const int lc = lane & 15;
  const int hf = lane >> 4;
#pragma unroll
  for (int nt = 0; nt < 4; ++nt) {
    int n = n0 + nt * 16 + lc;
    float bia = bias[n];
    int head = n >> 6, d = n & 63;
#pragma unroll
    for (int e = 0; e < 8; ++e) {
      int m = m0 + e + (hf << 3);
      int bb = m >> 12;
      int t  = m & (TT - 1);
      unsigned short hv = f2bf(acc[nt][e] + bia);
      if (z == 2) {
        vt[(((size_t)bb * HEADS + head) * DH + d) * TT + t] = hv;
      } else {
        unsigned short* dst = (z == 0) ? qh : kh;
        dst[(((size_t)bb * HEADS + head) * TT + t) * DH + d] = hv;
      }
    }
  }
}

// ---------------- Flash attention: async-LDS staged K/V, double buffered ----------------
// K tile: 32 keys x 64 dh, row stride 72 ushorts (144B). V tile: 64 dh x 32 keys,
// row stride 40 ushorts (80B). All waves iterate the block-uniform causal range.
#define KSTR 72
#define VSTR 40
__global__ __launch_bounds__(128) void flash_attn_kernel(
    const unsigned short* __restrict__ qh, const unsigned short* __restrict__ kh,
    const unsigned short* __restrict__ vt, unsigned short* __restrict__ attnb) {
  __shared__ unsigned short kt[2][32 * KSTR];
  __shared__ unsigned short vs[2][64 * VSTR];
  __shared__ unsigned short pls[4][16 * VSTR];

  const int tid  = threadIdx.x;
  const int lane = tid & 31;
  const int wid  = tid >> 5;
  const int lc   = lane & 15;
  const int hf   = lane >> 4;
  const int b    = blockIdx.z, h = blockIdx.y;
  const int q0   = blockIdx.x * 64 + wid * 16;
  const int kbmax = blockIdx.x * 64 + 64;      // uniform causal bound for the block

  const unsigned short* Q = qh + ((size_t)b * HEADS + h) * TT * DH;
  const unsigned short* K = kh + ((size_t)b * HEADS + h) * TT * DH;
  const unsigned short* V = vt + ((size_t)b * HEADS + h) * DH * TT;

  Frag qa0 = gfrag(Q, DH, q0, 0, lane);
  Frag qa1 = gfrag(Q, DH, q0, 32, lane);

  float mrow[8], lrow[8];
  v8f oacc[4];
#pragma unroll
  for (int e = 0; e < 8; ++e) { mrow[e] = -3.0e38f; lrow[e] = 0.f; }
#pragma unroll
  for (int nt = 0; nt < 4; ++nt) oacc[nt] = vzero8();

  unsigned short* myp = &pls[wid][0];
  const float SCALE = 0.125f;
  const float L2E   = 1.4426950408889634f;

  // stage key-block kb into buffer sel: 128 threads x 2 chunks each per tile
  auto stage = [&](int kb, int sel) {
    unsigned kbase = (unsigned)(size_t)&kt[sel][0];
    unsigned vbase = (unsigned)(size_t)&vs[sel][0];
#pragma unroll
    for (int i = 0; i < 2; ++i) {
      int c  = tid + i * 128;
      int kr = c >> 3, kc = c & 7;          // 32 rows x 8 x 16B
      async_b128(kbase + kr * (KSTR * 2) + kc * 16,
                 K + (size_t)(kb + kr) * DH + kc * 8);
      int vr = c >> 2, vc = c & 3;          // 64 rows x 4 x 16B
      async_b128(vbase + vr * (VSTR * 2) + vc * 16,
                 V + (size_t)vr * TT + kb + vc * 8);
    }
  };

  stage(0, 0);
  asm volatile("s_wait_asynccnt 0" ::: "memory");
  __syncthreads();

  int cur = 0;
  for (int kb = 0; kb < kbmax; kb += 32) {
    if (kb + 32 < kbmax) stage(kb + 32, cur ^ 1);   // overlap DMA with compute

    const unsigned short* ktc = &kt[cur][0];
    const unsigned short* vtc = &vs[cur][0];

    float sc[2][8];
#pragma unroll
    for (int nt = 0; nt < 2; ++nt) {
      Frag kf0 = lfrag(ktc, KSTR, nt * 16, 0, lane);
      Frag kf1 = lfrag(ktc, KSTR, nt * 16, 32, lane);
      v8f sa = vzero8();
      sa = WMMA_BF16(qa0.v, kf0.v, sa);
      sa = WMMA_BF16(qa1.v, kf1.v, sa);
      int col = kb + nt * 16 + lc;
#pragma unroll
      for (int e = 0; e < 8; ++e) {
        int row = q0 + e + (hf << 3);
        sc[nt][e] = (col > row) ? -3.0e38f : sa[e] * SCALE;
      }
    }
    // online softmax (rows live across the 16 lanes of each half-wave)
#pragma unroll
    for (int e = 0; e < 8; ++e) {
      float tmax = fmaxf(sc[0][e], sc[1][e]);
#pragma unroll
      for (int off = 8; off > 0; off >>= 1) tmax = fmaxf(tmax, __shfl_xor(tmax, off, 16));
      float mn   = fmaxf(mrow[e], tmax);
      float corr = __builtin_amdgcn_exp2f((mrow[e] - mn) * L2E);
      mrow[e] = mn;
      float p0 = __builtin_amdgcn_exp2f((sc[0][e] - mn) * L2E);
      float p1 = __builtin_amdgcn_exp2f((sc[1][e] - mn) * L2E);
      sc[0][e] = p0; sc[1][e] = p1;
      float rs = p0 + p1;
#pragma unroll
      for (int off = 8; off > 0; off >>= 1) rs += __shfl_xor(rs, off, 16);
      lrow[e] = lrow[e] * corr + rs;
#pragma unroll
      for (int nt = 0; nt < 4; ++nt) oacc[nt][e] *= corr;
    }
    // P (accumulator layout) -> per-wave LDS tile -> A-fragment
#pragma unroll
    for (int nt = 0; nt < 2; ++nt)
#pragma unroll
      for (int e = 0; e < 8; ++e)
        myp[(e + (hf << 3)) * VSTR + nt * 16 + lc] = f2bf(sc[nt][e]);
    asm volatile("s_wait_dscnt 0" ::: "memory");
    Frag pa = lfrag(myp, VSTR, 0, 0, lane);
    // O += P * V  (V^T tile resident in LDS)
#pragma unroll
    for (int nt = 0; nt < 4; ++nt) {
      Frag vf = lfrag(vtc, VSTR, nt * 16, 0, lane);
      oacc[nt] = WMMA_BF16(pa.v, vf.v, oacc[nt]);
    }

    asm volatile("s_wait_asynccnt 0" ::: "memory");  // next tiles landed
    __syncthreads();                                  // everyone done with cur
    cur ^= 1;
  }

  // normalize + store context bf16 [B, T, H*dh]
#pragma unroll
  for (int nt = 0; nt < 4; ++nt)
#pragma unroll
    for (int e = 0; e < 8; ++e) {
      int t  = q0 + e + (hf << 3);
      float o = oacc[nt][e] / lrow[e];
      attnb[((size_t)b * TT + t) * DM + h * DH + nt * 16 + lc] = f2bf(o);
    }
}

// ---------------- output projection: NT GEMM -> fp32, register double-buffered ----------------
__global__ __launch_bounds__(128) void oproj_kernel(
    const unsigned short* __restrict__ ab, const unsigned short* __restrict__ woT,
    const float* __restrict__ bo, float* __restrict__ out) {
  const int lane = threadIdx.x & 31;
  const int wid  = threadIdx.x >> 5;
  const int m0   = blockIdx.x * 64 + wid * 16;
  const int n0   = blockIdx.y * 64;

  v8f acc[4];
#pragma unroll
  for (int nt = 0; nt < 4; ++nt) acc[nt] = vzero8();

  Frag a = gfrag(ab, DM, m0, 0, lane);
  Frag b[4];
#pragma unroll
  for (int nt = 0; nt < 4; ++nt) b[nt] = gfrag(woT, DM, n0 + nt * 16, 0, lane);

#pragma unroll 4
  for (int k = 0; k < DM; k += 32) {
    int kn = (k + 32 < DM) ? (k + 32) : 0;
    Frag a2 = gfrag(ab, DM, m0, kn, lane);
    Frag b2[4];
#pragma unroll
    for (int nt = 0; nt < 4; ++nt) b2[nt] = gfrag(woT, DM, n0 + nt * 16, kn, lane);
#pragma unroll
    for (int nt = 0; nt < 4; ++nt) acc[nt] = WMMA_BF16(a.v, b[nt].v, acc[nt]);
    a = a2;
#pragma unroll
    for (int nt = 0; nt < 4; ++nt) b[nt] = b2[nt];
  }

  const int lc = lane & 15;
  const int hf = lane >> 4;
#pragma unroll
  for (int nt = 0; nt < 4; ++nt) {
    int n = n0 + nt * 16 + lc;
    float bia = bo[n];
#pragma unroll
    for (int e = 0; e < 8; ++e) {
      int m = m0 + e + (hf << 3);
      out[(size_t)m * DM + n] = acc[nt][e] + bia;
    }
  }
}

extern "C" void kernel_launch(void* const* d_in, const int* in_sizes, int n_in,
                              void* d_out, int out_size, void* d_ws, size_t ws_size,
                              hipStream_t stream) {
  const float* x  = (const float*)d_in[0];
  // d_in[1] = attn_mask (causal; implemented analytically in-kernel)
  const float* wq = (const float*)d_in[2];
  const float* bq = (const float*)d_in[3];
  const float* wk = (const float*)d_in[4];
  const float* bk = (const float*)d_in[5];
  const float* wv = (const float*)d_in[6];
  const float* bv = (const float*)d_in[7];
  const float* wo = (const float*)d_in[8];
  const float* bo = (const float*)d_in[9];
  float* out = (float*)d_out;

  char* ws = (char*)d_ws;
  const size_t SZ_X = (size_t)MTOT * DM * 2;
  const size_t SZ_W = (size_t)DM * DM * 2;
  unsigned short* xb  = (unsigned short*)(ws);
  unsigned short* qhb = (unsigned short*)(ws + 1 * SZ_X);
  unsigned short* khb = (unsigned short*)(ws + 2 * SZ_X);
  unsigned short* vtb = (unsigned short*)(ws + 3 * SZ_X);
  unsigned short* atb = (unsigned short*)(ws + 4 * SZ_X);
  unsigned short* wqT = (unsigned short*)(ws + 5 * SZ_X);
  unsigned short* wkT = (unsigned short*)(ws + 5 * SZ_X + 1 * SZ_W);
  unsigned short* wvT = (unsigned short*)(ws + 5 * SZ_X + 2 * SZ_W);
  unsigned short* woT = (unsigned short*)(ws + 5 * SZ_X + 3 * SZ_W);

  int nX = MTOT * DM;
  cvt_kernel<<<(nX + 255) / 256, 256, 0, stream>>>(x, xb, nX);

  dim3 tg(DM / 16, DM / 16);
  wtrans_kernel<<<tg, dim3(16, 16), 0, stream>>>(wq, wqT);
  wtrans_kernel<<<tg, dim3(16, 16), 0, stream>>>(wk, wkT);
  wtrans_kernel<<<tg, dim3(16, 16), 0, stream>>>(wv, wvT);
  wtrans_kernel<<<tg, dim3(16, 16), 0, stream>>>(wo, woT);

  qkv_gemm_kernel<<<dim3(MTOT / 64, DM / 64, 3), 128, 0, stream>>>(
      xb, wqT, wkT, wvT, bq, bk, bv, qhb, khb, vtb);

  flash_attn_kernel<<<dim3(TT / 64, HEADS, BB), 128, 0, stream>>>(qhb, khb, vtb, atb);

  oproj_kernel<<<dim3(MTOT / 64, DM / 64), 128, 0, stream>>>(atb, woT, bo, out);

  (void)in_sizes; (void)n_in; (void)out_size; (void)ws_size;
}
